// Attention_62620623176132
// MI455X (gfx1250) — matrix operand order repaired and checked by reading, not compiled, not run on hardware
//
#include <hip/hip_runtime.h>

typedef __attribute__((ext_vector_type(16))) __bf16 v16bf;
typedef __attribute__((ext_vector_type(8)))  float  v8f;
typedef __attribute__((ext_vector_type(4)))  unsigned int u32x4;
typedef __attribute__((ext_vector_type(8)))  int i32x8;
typedef __attribute__((ext_vector_type(4)))  int i32x4;

static __device__ __forceinline__ v8f wmma_bf16(v16bf a, v16bf b, v8f c) {
  return __builtin_amdgcn_wmma_f32_16x16x32_bf16(false, a, false, b, (short)0, c, false, false);
}

// low 32 bits of a flat pointer into the LDS aperture == logical LDS byte address
static __device__ __forceinline__ unsigned lds_off(const void* p) {
  return (unsigned)(unsigned long long)p;
}

// ---- Tensor Data Mover: 2D tile (bf16) global -> LDS, per ISA D# layout ----
// tile: tile_d0 elements (contiguous) x tile_d1 rows; row stride = stride_e elems
static __device__ __forceinline__ void tdm_load_2d(const void* gaddr, unsigned lds_addr,
                                                   unsigned tile_d0, unsigned tile_d1,
                                                   unsigned tens_d0, unsigned tens_d1,
                                                   unsigned stride_e) {
  unsigned long long ga = (unsigned long long)gaddr;
  u32x4 g0;
  g0[0] = 1u;                                   // count=1 (valid), user mode, no gather
  g0[1] = lds_addr;                             // D#.lds_addr (bytes)
  g0[2] = (unsigned)(ga & 0xFFFFFFFFull);       // global_addr[31:0]
  g0[3] = (unsigned)((ga >> 32) & 0x01FFFFFFull) | (2u << 30);  // [56:32] | type=2
  i32x8 g1;
  g1[0] = (int)(1u << 16);                      // workgroup_mask=0, data_size=1 (2B)
  g1[1] = (int)((tens_d0 & 0xFFFFu) << 16);     // tensor_dim0[15:0] @ bits 63:48
  g1[2] = (int)((tens_d0 >> 16) | ((tens_d1 & 0xFFFFu) << 16)); // dim0 hi | dim1 lo
  g1[3] = (int)((tens_d1 >> 16) | (tile_d0 << 16));             // dim1 hi | tile_dim0
  g1[4] = (int)(tile_d1 & 0xFFFFu);             // tile_dim1 (tile_dim2=0)
  g1[5] = (int)stride_e;                        // tensor_dim0_stride[31:0]
  g1[6] = 0;                                    // stride hi16 | dim1_stride lo16
  g1[7] = 0;
  i32x4 z4 = {0, 0, 0, 0};
  i32x8 z8 = {0, 0, 0, 0, 0, 0, 0, 0};
  __builtin_amdgcn_tensor_load_to_lds(g0, g1, z4, z4, z8, 0);
}

// ---- fragment loaders (bf16, per CDNA5 ISA VGPR layouts) -------------------
// A fragment 16x32 (MxK): lane l16 = row M; VGPR v holds K pair
//   v<4 : K = 2v + 8*half ; v>=4 : K = 16 + 2(v-4) + 8*half
static __device__ __forceinline__ v16bf load_a_frag(const __bf16* p, int ld,
                                                    unsigned l16, unsigned half) {
  union { v16bf v; unsigned u[8]; } f;
  const __bf16* row = p + (size_t)l16 * ld;
#pragma unroll
  for (int v = 0; v < 8; ++v) {
    unsigned kb = (v < 4) ? (2u * v + 8u * half) : (8u + 2u * v + 8u * half);
    f.u[v] = *(const unsigned*)(row + kb);
  }
  return f.v;
}

// B fragment 32x16 (KxN): lane l16 = col N; VGPR v holds K pair (same as A):
//   v<4 : K = 2v + 8*half ; v>=4 : K = 16 + 2(v-4) + 8*half
// Requirement: for column n, K elements contiguous: addr = p + n*ld + k
static __device__ __forceinline__ v16bf load_b_frag(const __bf16* p, int ld,
                                                    unsigned l16, unsigned half) {
  union { v16bf v; unsigned u[8]; } f;
  const __bf16* col = p + (size_t)l16 * ld;
#pragma unroll
  for (int v = 0; v < 8; ++v) {
    unsigned kb = (v < 4) ? (2u * v + 8u * half) : (8u + 2u * v + 8u * half);
    f.u[v] = *(const unsigned*)(col + kb);
  }
  return f.v;
}

// ---- elementwise casts -----------------------------------------------------
__global__ void cast_f32_bf16(const float* __restrict__ in, __bf16* __restrict__ out, int n) {
  int i = blockIdx.x * blockDim.x + threadIdx.x;
  if (i < n) out[i] = (__bf16)in[i];
}

// wt[n*K + k] = w[k*N + n]   (B-fragment K-pairs become contiguous)
__global__ void transpose_cast_bf16(const float* __restrict__ w, __bf16* __restrict__ wt,
                                    int N, int K) {
  int i = blockIdx.x * blockDim.x + threadIdx.x;
  if (i >= N * K) return;
  int n = i / K, k = i - n * K;
  wt[(size_t)n * K + k] = (__bf16)w[(size_t)k * N + n];
}

// ---- TDM-fed, double-buffered WMMA GEMM: C[M,N] = A[M,K] * Wt[N,K]^T ------
#define TM 128
#define TN 64
#define TK 32
template <bool BF16OUT>
__global__ void __launch_bounds__(256)
gemm_bf16_wmma(const __bf16* __restrict__ A, const __bf16* __restrict__ Bt,
               void* __restrict__ Cout, int M, int N, int K) {
  __shared__ __bf16 As[2][TM * TK];
  __shared__ __bf16 Bs[2][TN * TK];
  const int tid = threadIdx.x;
  const unsigned lane = tid & 31u, half = lane >> 4, l16 = lane & 15u;
  const int wv = tid >> 5;                  // 0..7
  const int rw = wv >> 1, cw = wv & 1;
  const int m0 = blockIdx.y * TM, n0 = blockIdx.x * TN;
  v8f acc[2][2] = {};

  const __bf16* Arow = A + (size_t)m0 * K;
  const __bf16* Brow = Bt + (size_t)n0 * K;
  const int iters = K / TK;

  if (wv == 0) {  // prologue: DMA tile 0 into buffer 0 (TDM, async, TENSORcnt)
    tdm_load_2d(Arow, lds_off(&As[0][0]), TK, TM, (unsigned)K, (unsigned)(M - m0), (unsigned)K);
    tdm_load_2d(Brow, lds_off(&Bs[0][0]), TK, TN, (unsigned)K, (unsigned)(N - n0), (unsigned)K);
  }
  for (int it = 0; it < iters; ++it) {
    const int buf = it & 1;
    if (wv == 0) {
      if (it + 1 < iters) {  // prefetch next tile into the other buffer
        tdm_load_2d(Arow + (it + 1) * TK, lds_off(&As[buf ^ 1][0]), TK, TM,
                    (unsigned)(K - (it + 1) * TK), (unsigned)(M - m0), (unsigned)K);
        tdm_load_2d(Brow + (it + 1) * TK, lds_off(&Bs[buf ^ 1][0]), TK, TN,
                    (unsigned)(K - (it + 1) * TK), (unsigned)(N - n0), (unsigned)K);
        __builtin_amdgcn_s_wait_tensorcnt((short)2);  // tile `it` landed
      } else {
        __builtin_amdgcn_s_wait_tensorcnt((short)0);
      }
    }
    __syncthreads();  // publish tile `it` to all waves
    v16bf af[2], bfr[2];
#pragma unroll
    for (int i = 0; i < 2; ++i) af[i]  = load_a_frag(&As[buf][(32 * rw + 16 * i) * TK], TK, l16, half);
#pragma unroll
    for (int j = 0; j < 2; ++j) bfr[j] = load_b_frag(&Bs[buf][(32 * cw + 16 * j) * TK], TK, l16, half);
#pragma unroll
    for (int i = 0; i < 2; ++i)
#pragma unroll
      for (int j = 0; j < 2; ++j)
        acc[i][j] = wmma_bf16(af[i], bfr[j], acc[i][j]);
    __syncthreads();  // all reads of buf done before TDM may overwrite it next iter
  }
  // epilogue: C layout VGPR e -> row e + 8*half, lane -> col
#pragma unroll
  for (int i = 0; i < 2; ++i)
#pragma unroll
    for (int j = 0; j < 2; ++j)
#pragma unroll
      for (int e = 0; e < 8; ++e) {
        int row = m0 + 32 * rw + 16 * i + e + 8 * (int)half;
        int col = n0 + 32 * cw + 16 * j + (int)l16;
        float val = acc[i][j][e];
        if constexpr (BF16OUT) ((__bf16*)Cout)[(size_t)row * N + col] = (__bf16)val;
        else                   ((float*)Cout)[(size_t)row * N + col] = val;
      }
}

// ---- RoPE + relayout: token-major QKV -> head-major Q/K, transposed V ------
__global__ void rope_layout(const __bf16* __restrict__ Qt, const __bf16* __restrict__ Kt,
                            const __bf16* __restrict__ Vtk,
                            const float* __restrict__ cosf, const float* __restrict__ sinf,
                            __bf16* __restrict__ Qh, __bf16* __restrict__ Kh,
                            __bf16* __restrict__ Vt, int total) {
  constexpr int S = 2048, H = 16, HD = 128, D = 2048, HP = HD / 2;
  int idx = blockIdx.x * blockDim.x + threadIdx.x;
  if (idx >= total) return;
  int dp = idx & 63;
  int h  = (idx >> 6) & 15;
  int s  = (idx >> 10) & (S - 1);
  int b  = idx >> 21;
  size_t tok = ((size_t)(b * S + s)) * D + h * HD + 2 * dp;
  float c  = cosf[s * HP + dp], sn = sinf[s * HP + dp];
  float qr = (float)Qt[tok], qi = (float)Qt[tok + 1];
  float kr = (float)Kt[tok], ki = (float)Kt[tok + 1];
  int bh = b * H + h;
  size_t hm = ((size_t)bh * S + s) * HD + 2 * dp;
  Qh[hm]     = (__bf16)(qr * c - qi * sn);
  Qh[hm + 1] = (__bf16)(qr * sn + qi * c);
  Kh[hm]     = (__bf16)(kr * c - ki * sn);
  Kh[hm + 1] = (__bf16)(kr * sn + ki * c);
  size_t vt = ((size_t)bh * HD + 2 * dp) * S + s;
  Vt[vt]     = Vtk[tok];
  Vt[vt + S] = Vtk[tok + 1];
}

// ---- flash attention: grid(S/64, B*H), 4 waves x 16 q-rows, TDM K/V tiles --
__global__ void __launch_bounds__(128)
flash_attn(const __bf16* __restrict__ Q, const __bf16* __restrict__ Kh,
           const __bf16* __restrict__ Vt, __bf16* __restrict__ O) {
  constexpr int S = 2048, HD = 128, H = 16, D = 2048;
  const float scale = 0.08838834764831845f;  // 1/sqrt(128)
  __shared__ __bf16 Ks[32 * HD];   // [key][d]   8KB
  __shared__ __bf16 Vs[HD * 32];   // [d][krel]  8KB
  __shared__ __bf16 Pl[4][16 * 32];

  const int w = threadIdx.x >> 5;
  const unsigned lane = threadIdx.x & 31u, half = lane >> 4, l16 = lane & 15u;
  const int bh = blockIdx.y, b = bh >> 4, h = bh & 15;
  const int qblk = blockIdx.x * 64, q0 = qblk + 16 * w;

  const __bf16* qp = Q  + ((size_t)bh * S + q0) * HD;
  const __bf16* kp = Kh + (size_t)bh * S * HD;
  const __bf16* vp = Vt + (size_t)bh * HD * S;

  // pre-scaled Q A-fragments (4 chunks of K=32 over HD=128)
  v16bf qa[4];
#pragma unroll
  for (int c = 0; c < 4; ++c) {
    union { v16bf v; __bf16 e[16]; } f;
    const __bf16* row = qp + (size_t)l16 * HD + 32 * c;
#pragma unroll
    for (int v = 0; v < 8; ++v) {
      unsigned kb = (v < 4) ? (2u * v + 8u * half) : (8u + 2u * v + 8u * half);
      f.e[2 * v]     = (__bf16)((float)row[kb]     * scale);
      f.e[2 * v + 1] = (__bf16)((float)row[kb + 1] * scale);
    }
    qa[c] = f.v;
  }

  v8f o[8] = {};
  float mrow[8], lrow[8];
#pragma unroll
  for (int i = 0; i < 8; ++i) { mrow[i] = -3.0e38f; lrow[i] = 0.0f; }

  const int kend = qblk + 64;  // causal bound, uniform across block
  for (int j0 = 0; j0 < kend; j0 += 32) {
    if (w == 0) {  // DMA K tile (32xHD) and V tile (HDx32) via TDM
      tdm_load_2d(kp + (size_t)j0 * HD, lds_off(&Ks[0]), HD, 32,
                  (unsigned)HD, (unsigned)(S - j0), (unsigned)HD);
      tdm_load_2d(vp + j0, lds_off(&Vs[0]), 32, HD,
                  (unsigned)(S - j0), (unsigned)HD, (unsigned)S);
      __builtin_amdgcn_s_wait_tensorcnt((short)0);
    }
    __syncthreads();  // K/V tiles visible

    v8f s0 = {}, s1 = {};
#pragma unroll
    for (int c = 0; c < 4; ++c) {
      v16bf bk0 = load_b_frag(&Ks[32 * c], HD, l16, half);
      v16bf bk1 = load_b_frag(&Ks[16 * HD + 32 * c], HD, l16, half);
      s0 = wmma_bf16(qa[c], bk0, s0);
      s1 = wmma_bf16(qa[c], bk1, s1);
    }
    // online softmax (score rows live across lanes 0..15 of each half)
#pragma unroll
    for (int i = 0; i < 8; ++i) {
      int qg  = q0 + i + 8 * (int)half;
      int kg0 = j0 + (int)l16, kg1 = kg0 + 16;
      float v0 = s0[i] + (kg0 > qg ? -1.0e9f : 0.0f);
      float v1 = s1[i] + (kg1 > qg ? -1.0e9f : 0.0f);
      float r = fmaxf(v0, v1);
      r = fmaxf(r, __shfl_xor(r, 1, 32));
      r = fmaxf(r, __shfl_xor(r, 2, 32));
      r = fmaxf(r, __shfl_xor(r, 4, 32));
      r = fmaxf(r, __shfl_xor(r, 8, 32));
      float mn = fmaxf(mrow[i], r);
      float alpha = __expf(mrow[i] - mn);
      float p0 = __expf(v0 - mn), p1 = __expf(v1 - mn);
      float rs = p0 + p1;
      rs += __shfl_xor(rs, 1, 32);
      rs += __shfl_xor(rs, 2, 32);
      rs += __shfl_xor(rs, 4, 32);
      rs += __shfl_xor(rs, 8, 32);
      lrow[i] = lrow[i] * alpha + rs;
      mrow[i] = mn;
      s0[i] = p0; s1[i] = p1;
#pragma unroll
      for (int dt = 0; dt < 8; ++dt) o[dt][i] *= alpha;
    }
    // C-layout P tile -> LDS -> A-layout fragment
#pragma unroll
    for (int i = 0; i < 8; ++i) {
      int qr = i + 8 * (int)half;
      Pl[w][qr * 32 + (int)l16]      = (__bf16)s0[i];
      Pl[w][qr * 32 + 16 + (int)l16] = (__bf16)s1[i];
    }
    __syncthreads();  // P ready; also all K-tile reads done
    v16bf pa = load_a_frag(Pl[w], 32, l16, half);
#pragma unroll
    for (int dt = 0; dt < 8; ++dt) {
      v16bf bv = load_b_frag(&Vs[16 * dt * 32], 32, l16, half);
      o[dt] = wmma_bf16(pa, bv, o[dt]);
    }
    __syncthreads();  // all V-tile reads done before next TDM overwrites
  }
  // normalize + store token-major bf16 [B,S,H*HD]
#pragma unroll
  for (int i = 0; i < 8; ++i) {
    float inv = 1.0f / lrow[i];
    int srow = q0 + i + 8 * (int)half;
    size_t base = ((size_t)b * S + srow) * D + h * HD;
#pragma unroll
    for (int dt = 0; dt < 8; ++dt)
      O[base + 16 * dt + (int)l16] = (__bf16)(o[dt][i] * inv);
  }
}

// ---- host orchestration ----------------------------------------------------
extern "C" void kernel_launch(void* const* d_in, const int* in_sizes, int n_in,
                              void* d_out, int out_size, void* d_ws, size_t ws_size,
                              hipStream_t stream) {
  constexpr int B = 2, S = 2048, D = 2048, H = 16, HD = 128;
  constexpr int M = B * S, K = D, N = D;
  const float* x  = (const float*)d_in[0];
  const float* wq = (const float*)d_in[1];
  const float* wk = (const float*)d_in[2];
  const float* wv = (const float*)d_in[3];
  const float* wo = (const float*)d_in[4];
  const float* fc = (const float*)d_in[5];
  const float* fs = (const float*)d_in[6];
  float* out = (float*)d_out;

  size_t off = 0;
  auto alloc = [&](size_t bytes) -> void* {
    void* p = (char*)d_ws + off;
    off += (bytes + 255) & ~(size_t)255;
    return p;
  };
  __bf16* xb   = (__bf16*)alloc((size_t)M * K * 2);
  __bf16* wqt  = (__bf16*)alloc((size_t)N * K * 2);
  __bf16* wkt  = (__bf16*)alloc((size_t)N * K * 2);
  __bf16* wvt  = (__bf16*)alloc((size_t)N * K * 2);
  __bf16* wot  = (__bf16*)alloc((size_t)N * K * 2);
  __bf16* qtok = (__bf16*)alloc((size_t)M * N * 2);
  __bf16* ktok = (__bf16*)alloc((size_t)M * N * 2);
  __bf16* vtok = (__bf16*)alloc((size_t)M * N * 2);
  __bf16* qh   = (__bf16*)alloc((size_t)M * N * 2);
  __bf16* kh   = (__bf16*)alloc((size_t)M * N * 2);
  __bf16* vT   = (__bf16*)alloc((size_t)M * N * 2);
  __bf16* attnb = qtok;  // reuse: qtok dead after rope_layout

  int nx = M * K;
  cast_f32_bf16<<<(nx + 255) / 256, 256, 0, stream>>>(x, xb, nx);
  int nw = N * K;
  transpose_cast_bf16<<<(nw + 255) / 256, 256, 0, stream>>>(wq, wqt, N, K);
  transpose_cast_bf16<<<(nw + 255) / 256, 256, 0, stream>>>(wk, wkt, N, K);
  transpose_cast_bf16<<<(nw + 255) / 256, 256, 0, stream>>>(wv, wvt, N, K);
  transpose_cast_bf16<<<(nw + 255) / 256, 256, 0, stream>>>(wo, wot, N, K);

  dim3 gg(N / TN, M / TM);
  gemm_bf16_wmma<true><<<gg, 256, 0, stream>>>(xb, wqt, qtok, M, N, K);
  gemm_bf16_wmma<true><<<gg, 256, 0, stream>>>(xb, wkt, ktok, M, N, K);
  gemm_bf16_wmma<true><<<gg, 256, 0, stream>>>(xb, wvt, vtok, M, N, K);

  int nr = B * S * H * (HD / 2);
  rope_layout<<<(nr + 255) / 256, 256, 0, stream>>>(qtok, ktok, vtok, fc, fs, qh, kh, vT, nr);

  flash_attn<<<dim3(S / 64, B * H), 128, 0, stream>>>(qh, kh, vT, attnb);

  gemm_bf16_wmma<false><<<gg, 256, 0, stream>>>(attnb, wot, out, M, N, K);
  (void)in_sizes; (void)n_in; (void)out_size; (void)ws_size;
}
